// DecoderLayer_14611478741087
// MI455X (gfx1250) — matrix operand order repaired and checked by reading, not compiled, not run on hardware
//
#include <hip/hip_runtime.h>

// Problem constants (match reference)
#define B_  4
#define T_  2048
#define S_  2048
#define D_  256
#define F_  1024
#define M_  (B_ * T_)        // 8192 rows of activations
#define EPS_ 1e-5f

// GEMM block tiling
#define BM_ 64
#define BN_ 64
#define KB_ 32

typedef unsigned short u16;
typedef __attribute__((ext_vector_type(16))) __bf16       v16bf;
typedef __attribute__((ext_vector_type(8)))  float        v8f;
typedef __attribute__((ext_vector_type(4)))  unsigned int u32x4;

union Frag16 { u32x4 u[2]; v16bf v; };

__device__ __forceinline__ u16 f2bf(float f) {
  union { float f; unsigned u; } x; x.f = f;
  unsigned r = x.u + 0x7FFFu + ((x.u >> 16) & 1u);   // round-to-nearest-even
  return (u16)(r >> 16);
}

__device__ __forceinline__ v8f zero8() {
  v8f z = {0.f, 0.f, 0.f, 0.f, 0.f, 0.f, 0.f, 0.f};
  return z;
}

// A-matrix 16x32 bf16 fragment: lane<16 holds row=lane, K={0..15};
// lane>=16 holds row=lane-16, K={16..31}. Two contiguous 16B loads.
__device__ __forceinline__ v16bf load_a16(const u16* A, int lda, int m0, int k0, int lane) {
  int m = lane & 15, hi = lane >> 4;
  const u16* p = A + (size_t)(m0 + m) * lda + k0 + hi * 16;
  Frag16 f;
  f.u[0] = *(const u32x4*)(p);
  f.u[1] = *(const u32x4*)(p + 8);
  return f.v;
}

// B-matrix 32x16 bf16 fragment from a [N,K] row-major ("transposed") buffer:
// lane holds col n=lane&15, K = hi*16 + j (16 contiguous elements -> 32B).
__device__ __forceinline__ v16bf load_b16(const u16* Bt, int ldb, int n0, int k0, int lane) {
  int n = lane & 15, hi = lane >> 4;
  const u16* p = Bt + (size_t)(n0 + n) * ldb + k0 + hi * 16;
  Frag16 f;
  f.u[0] = *(const u32x4*)(p);
  f.u[1] = *(const u32x4*)(p + 8);
  return f.v;
}

__device__ __forceinline__ v8f wmma_bf16(v16bf a, v16bf b, v8f c) {
  return __builtin_amdgcn_wmma_f32_16x16x32_bf16(false, a, false, b, (short)0, c, false, false);
}

// CDNA5 async DMA: global -> LDS, 16B per lane, tracked by ASYNCcnt.
// lds operand is the addrspace(3) byte offset (low dword of the flat pointer).
__device__ __forceinline__ void async_ld128(const u16* gaddr, const u16* lds) {
  unsigned loff = (unsigned)(uintptr_t)lds;
  unsigned long long ga = (unsigned long long)(uintptr_t)gaddr;
  asm volatile("global_load_async_to_lds_b128 %0, %1, off"
               :: "v"(loff), "v"(ga) : "memory");
}
__device__ __forceinline__ void wait_async0() {
  asm volatile("s_wait_asynccnt 0x0" ::: "memory");
}

// ---------------------------------------------------------------- cast kernels
__global__ void cast_bf16_kernel(const float* __restrict__ in, u16* __restrict__ out, int n) {
  int i = blockIdx.x * blockDim.x + threadIdx.x;
  if (i < n) out[i] = f2bf(in[i]);
}

// W[K,N] f32 -> Wt[N,K] bf16
__global__ void cast_transpose_kernel(const float* __restrict__ in, u16* __restrict__ out,
                                      int K, int N) {
  int i = blockIdx.x * blockDim.x + threadIdx.x;
  if (i < K * N) {
    int k = i / N, n = i % N;
    out[(size_t)n * K + k] = f2bf(in[i]);
  }
}

// ---------------------------------------------------------------- WMMA GEMM
// C[M,N] = A_bf16[M,K] @ Bt_bf16[N,K]^T + bias.
// 64x64 block tile, 4 waves; A/B staged in LDS with double-buffered
// global_load_async_to_lds_b128 (ASYNCcnt); one barrier per k-step; DMA of
// stage k+1 overlapped with WMMA of stage k.  All B fragments hoisted ahead of
// the WMMA chain so the compiler can use partial s_wait_dscnt.
// mode: 0=f32 [M,N], 1=bf16 [M,N], 2=bf16 transposed-per-batch.
__global__ void gemm_wmma_kernel(const u16* __restrict__ A, const u16* __restrict__ Bt,
                                 const float* __restrict__ bias,
                                 float* __restrict__ outF, u16* __restrict__ outBf,
                                 int M, int N, int K, int mode, int relu,
                                 int rows_per_batch) {
  __shared__ __align__(16) u16 Abuf[2][BM_][KB_];   // 8 KB
  __shared__ __align__(16) u16 Bbuf[2][BN_][KB_];   // 8 KB
  int lane = threadIdx.x;
  int wave = threadIdx.y;
  int tid  = wave * 32 + lane;          // 0..127
  int mblk = blockIdx.y * BM_;
  int nblk = blockIdx.x * BN_;

  // staging assignment: thread -> (row r, 32B half h), two b128 per matrix
  int r = tid >> 1;                      // 0..63
  int h = (tid & 1) * 16;                // element offset 0 or 16

  auto stage_issue = [&](int s) {
    int buf = s & 1;
    int k0  = s * KB_;
    const u16* gA = A  + (size_t)(mblk + r) * K + k0 + h;
    const u16* gB = Bt + (size_t)(nblk + r) * K + k0 + h;
    async_ld128(gA,     &Abuf[buf][r][h]);
    async_ld128(gA + 8, &Abuf[buf][r][h + 8]);
    async_ld128(gB,     &Bbuf[buf][r][h]);
    async_ld128(gB + 8, &Bbuf[buf][r][h + 8]);
  };

  v8f acc[4];
#pragma unroll
  for (int i = 0; i < 4; ++i) acc[i] = zero8();

  int nk = K / KB_;
  stage_issue(0);
  for (int ks = 0; ks < nk; ++ks) {
    wait_async0();                       // this wave's stage-ks DMA done
    // single barrier: proves every wave finished compute(ks-1) (program order)
    // before anyone overwrites buffer (ks+1)%2 below.
    __syncthreads();
    if (ks + 1 < nk) stage_issue(ks + 1);  // overlap next DMA with compute
    int buf = ks & 1;
    // hoist all LDS fragment loads ahead of the WMMA chain
    v16bf a  = load_a16(&Abuf[buf][wave * 16][0], KB_, 0, 0, lane);
    v16bf b0 = load_b16(&Bbuf[buf][0][0], KB_, 0,  0, lane);
    v16bf b1 = load_b16(&Bbuf[buf][0][0], KB_, 16, 0, lane);
    v16bf b2 = load_b16(&Bbuf[buf][0][0], KB_, 32, 0, lane);
    v16bf b3 = load_b16(&Bbuf[buf][0][0], KB_, 48, 0, lane);
    acc[0] = wmma_bf16(a, b0, acc[0]);
    acc[1] = wmma_bf16(a, b1, acc[1]);
    acc[2] = wmma_bf16(a, b2, acc[2]);
    acc[3] = wmma_bf16(a, b3, acc[3]);
  }

  int n = lane & 15, hi = lane >> 4;
  int m0 = mblk + wave * 16;
#pragma unroll
  for (int nt = 0; nt < 4; ++nt) {
    int n0 = nblk + nt * 16;
    float bv = bias ? bias[n0 + n] : 0.f;
#pragma unroll
    for (int rr = 0; rr < 8; ++rr) {
      int m = m0 + rr + 8 * hi;
      float v = acc[nt][rr] + bv;
      if (relu) v = v > 0.f ? v : 0.f;
      if (mode == 0) {
        outF[(size_t)m * N + n0 + n] = v;
      } else if (mode == 1) {
        outBf[(size_t)m * N + n0 + n] = f2bf(v);
      } else {
        int bb = m / rows_per_batch, tt = m % rows_per_batch;
        outBf[(size_t)bb * N * rows_per_batch + (size_t)(n0 + n) * rows_per_batch + tt] =
            f2bf(v);
      }
    }
  }
}

// ---------------------------------------------------------------- flash attention
// Q bf16 [B,Tq,D], Kb bf16 [B,Sk,D], VT bf16 [B,D,Sk]; O f32 [B,Tq,D].
// One wave per 16-row Q tile; online softmax; P routed through LDS to become
// a bf16 A-fragment for the PV WMMA.
__global__ void flash_attn_wmma_kernel(const u16* __restrict__ Q, const u16* __restrict__ Kb,
                                       const u16* __restrict__ VT, float* __restrict__ O,
                                       int Tq, int Sk, int causal) {
  __shared__ __align__(16) u16 Pst[4][16 * 32];
  int lane = threadIdx.x;
  int wave = threadIdx.y;
  int tile = blockIdx.x * 4 + wave;
  int tilesPerBatch = Tq / 16;
  int b  = tile / tilesPerBatch;
  int m0 = (tile % tilesPerBatch) * 16;

  const u16* Qb  = Q  + (size_t)b * Tq * D_;
  const u16* Kbb = Kb + (size_t)b * Sk * D_;
  const u16* Vb  = VT + (size_t)b * D_ * Sk;
  u16* P = Pst[wave];
  int n = lane & 15, hi = lane >> 4;

  v8f o[D_ / 16];
#pragma unroll
  for (int i = 0; i < D_ / 16; ++i) o[i] = zero8();
  float mx[8], sm[8];
#pragma unroll
  for (int r = 0; r < 8; ++r) { mx[r] = -1e30f; sm[r] = 0.f; }

  const float scale = 0.0625f;  // 1/sqrt(256)
  int sEnd = causal ? (m0 + 16) : Sk;

  for (int s0 = 0; s0 < sEnd; s0 += 32) {
    v8f sc0 = zero8(), sc1 = zero8();
#pragma unroll
    for (int kk = 0; kk < D_ / 32; ++kk) {
      v16bf a  = load_a16(Qb, D_, m0, kk * 32, lane);
      v16bf b0 = load_b16(Kbb, D_, s0,      kk * 32, lane);
      v16bf b1 = load_b16(Kbb, D_, s0 + 16, kk * 32, lane);
      sc0 = wmma_bf16(a, b0, sc0);
      sc1 = wmma_bf16(a, b1, sc1);
    }
#pragma unroll
    for (int r = 0; r < 8; ++r) {
      int row = m0 + r + 8 * hi;
      float v0 = sc0[r] * scale;
      float v1 = sc1[r] * scale;
      if (causal) {
        if (s0 + n > row)      v0 = -1e30f;
        if (s0 + 16 + n > row) v1 = -1e30f;
      }
      float mrow = fmaxf(v0, v1);
#pragma unroll
      for (int xm = 8; xm >= 1; xm >>= 1) mrow = fmaxf(mrow, __shfl_xor(mrow, xm));
      float nm    = fmaxf(mx[r], mrow);
      float alpha = __expf(mx[r] - nm);
      mx[r] = nm;
      float p0 = __expf(v0 - nm);
      float p1 = __expf(v1 - nm);
      float ps = p0 + p1;
#pragma unroll
      for (int xm = 8; xm >= 1; xm >>= 1) ps += __shfl_xor(ps, xm);
      sm[r] = sm[r] * alpha + ps;
#pragma unroll
      for (int dt = 0; dt < D_ / 16; ++dt) o[dt][r] *= alpha;
      int rl = r + 8 * hi;
      P[rl * 32 + n]      = f2bf(p0);
      P[rl * 32 + 16 + n] = f2bf(p1);
    }
    // per-wave LDS RAW: DS ops are in-order per wave; fence compiler + DScnt
    asm volatile("s_wait_dscnt 0x0" ::: "memory");
    v16bf pf = load_a16(P, 32, 0, 0, lane);
#pragma unroll
    for (int dt = 0; dt < D_ / 16; ++dt) {
      v16bf bv = load_b16(Vb, Sk, dt * 16, s0, lane);
      o[dt] = wmma_bf16(pf, bv, o[dt]);
    }
  }

  float* Ob = O + (size_t)b * Tq * D_;
#pragma unroll
  for (int dt = 0; dt < D_ / 16; ++dt) {
#pragma unroll
    for (int r = 0; r < 8; ++r) {
      int m = m0 + r + 8 * hi;
      Ob[(size_t)m * D_ + dt * 16 + n] = o[dt][r] / sm[r];
    }
  }
}

// ---------------------------------------------------------------- residual + LayerNorm
__global__ void ln_residual_kernel(const float* __restrict__ x, const float* __restrict__ r,
                                   const float* __restrict__ g, const float* __restrict__ be,
                                   float* __restrict__ outF, u16* __restrict__ outBf) {
  __shared__ float red[D_];
  int row = blockIdx.x;
  int tid = threadIdx.x;   // 256 threads == D_
  size_t idx = (size_t)row * D_ + tid;
  float v = x[idx] + r[idx];

  red[tid] = v;
  __syncthreads();
  for (int s = D_ / 2; s > 0; s >>= 1) {
    if (tid < s) red[tid] += red[tid + s];
    __syncthreads();
  }
  float mean = red[0] * (1.0f / D_);
  __syncthreads();
  float d = v - mean;
  red[tid] = d * d;
  __syncthreads();
  for (int s = D_ / 2; s > 0; s >>= 1) {
    if (tid < s) red[tid] += red[tid + s];
    __syncthreads();
  }
  float var = red[0] * (1.0f / D_);
  float o = d * rsqrtf(var + EPS_) * g[tid] + be[tid];
  outF[idx] = o;
  if (outBf) outBf[idx] = f2bf(o);
}

// ---------------------------------------------------------------- launch
extern "C" void kernel_launch(void* const* d_in, const int* in_sizes, int n_in,
                              void* d_out, int out_size, void* d_ws, size_t ws_size,
                              hipStream_t stream) {
  (void)in_sizes; (void)n_in; (void)out_size; (void)ws_size;
  const float* y   = (const float*)d_in[0];
  const float* Z   = (const float*)d_in[1];
  // d_in[2] = target_mask (causal) -- implied by causal flag
  const float* Wq1 = (const float*)d_in[3];  const float* bq1 = (const float*)d_in[4];
  const float* Wk1 = (const float*)d_in[5];  const float* bk1 = (const float*)d_in[6];
  const float* Wv1 = (const float*)d_in[7];  const float* bv1 = (const float*)d_in[8];
  const float* Wq2 = (const float*)d_in[9];  const float* bq2 = (const float*)d_in[10];
  const float* Wk2 = (const float*)d_in[11]; const float* bk2 = (const float*)d_in[12];
  const float* Wv2 = (const float*)d_in[13]; const float* bv2 = (const float*)d_in[14];
  const float* W1  = (const float*)d_in[15]; const float* b1  = (const float*)d_in[16];
  const float* W2  = (const float*)d_in[17]; const float* b2  = (const float*)d_in[18];
  const float* g1  = (const float*)d_in[19]; const float* be1 = (const float*)d_in[20];
  const float* g2  = (const float*)d_in[21]; const float* be2 = (const float*)d_in[22];
  const float* g3  = (const float*)d_in[23]; const float* be3 = (const float*)d_in[24];
  float* out = (float*)d_out;

  char* w = (char*)d_ws;
  size_t off = 0;
  auto alloc = [&](size_t bytes) -> void* {
    void* p = w + off;
    off += (bytes + 255) & ~(size_t)255;
    return p;
  };

  u16* ybf  = (u16*)alloc((size_t)M_ * D_ * 2);
  u16* Zbf  = (u16*)alloc((size_t)B_ * S_ * D_ * 2);
  u16* Wq1T = (u16*)alloc((size_t)D_ * D_ * 2);
  u16* Wk1T = (u16*)alloc((size_t)D_ * D_ * 2);
  u16* Wv1T = (u16*)alloc((size_t)D_ * D_ * 2);
  u16* Wq2T = (u16*)alloc((size_t)D_ * D_ * 2);
  u16* Wk2T = (u16*)alloc((size_t)D_ * D_ * 2);
  u16* Wv2T = (u16*)alloc((size_t)D_ * D_ * 2);
  u16* W1T  = (u16*)alloc((size_t)F_ * D_ * 2);
  u16* W2T  = (u16*)alloc((size_t)D_ * F_ * 2);
  u16* Qb   = (u16*)alloc((size_t)M_ * D_ * 2);
  u16* Kbf  = (u16*)alloc((size_t)B_ * S_ * D_ * 2);
  u16* VT   = (u16*)alloc((size_t)B_ * D_ * S_ * 2);
  float* attn = (float*)alloc((size_t)M_ * D_ * 4);   // reused: sa, ca, ffn-out
  float* x1f  = (float*)alloc((size_t)M_ * D_ * 4);
  u16*   x1bf = (u16*)  alloc((size_t)M_ * D_ * 2);
  float* x2f  = (float*)alloc((size_t)M_ * D_ * 4);
  u16*   x2bf = (u16*)  alloc((size_t)M_ * D_ * 2);
  u16*   hbuf = (u16*)  alloc((size_t)M_ * F_ * 2);

  const dim3 gThr(32, 4);

  // 1) casts
  {
    int n = M_ * D_;
    cast_bf16_kernel<<<n / 256, 256, 0, stream>>>(y, ybf, n);
    cast_bf16_kernel<<<n / 256, 256, 0, stream>>>(Z, Zbf, n);
    int wdd = D_ * D_;
    cast_transpose_kernel<<<(wdd + 255) / 256, 256, 0, stream>>>(Wq1, Wq1T, D_, D_);
    cast_transpose_kernel<<<(wdd + 255) / 256, 256, 0, stream>>>(Wk1, Wk1T, D_, D_);
    cast_transpose_kernel<<<(wdd + 255) / 256, 256, 0, stream>>>(Wv1, Wv1T, D_, D_);
    cast_transpose_kernel<<<(wdd + 255) / 256, 256, 0, stream>>>(Wq2, Wq2T, D_, D_);
    cast_transpose_kernel<<<(wdd + 255) / 256, 256, 0, stream>>>(Wk2, Wk2T, D_, D_);
    cast_transpose_kernel<<<(wdd + 255) / 256, 256, 0, stream>>>(Wv2, Wv2T, D_, D_);
    int wdf = D_ * F_;
    cast_transpose_kernel<<<(wdf + 255) / 256, 256, 0, stream>>>(W1, W1T, D_, F_);
    cast_transpose_kernel<<<(wdf + 255) / 256, 256, 0, stream>>>(W2, W2T, F_, D_);
  }

  // 2) self-attention Q/K/V projections (V stored transposed per batch)
  gemm_wmma_kernel<<<dim3(D_ / BN_, M_ / BM_), gThr, 0, stream>>>(
      ybf, Wq1T, bq1, nullptr, Qb, M_, D_, D_, 1, 0, 0);
  gemm_wmma_kernel<<<dim3(D_ / BN_, M_ / BM_), gThr, 0, stream>>>(
      ybf, Wk1T, bk1, nullptr, Kbf, M_, D_, D_, 1, 0, 0);
  gemm_wmma_kernel<<<dim3(D_ / BN_, M_ / BM_), gThr, 0, stream>>>(
      ybf, Wv1T, bv1, nullptr, VT, M_, D_, D_, 2, 0, T_);

  // 3) causal flash attention
  flash_attn_wmma_kernel<<<(B_ * T_ / 16) / 4, gThr, 0, stream>>>(
      Qb, Kbf, VT, attn, T_, T_, 1);

  // 4) LN1: y1 = LN(y + sa)
  ln_residual_kernel<<<M_, D_, 0, stream>>>(y, attn, g1, be1, x1f, x1bf);

  // 5) cross-attention projections (K/V from Z)
  gemm_wmma_kernel<<<dim3(D_ / BN_, M_ / BM_), gThr, 0, stream>>>(
      x1bf, Wq2T, bq2, nullptr, Qb, M_, D_, D_, 1, 0, 0);
  gemm_wmma_kernel<<<dim3(D_ / BN_, (B_ * S_) / BM_), gThr, 0, stream>>>(
      Zbf, Wk2T, bk2, nullptr, Kbf, B_ * S_, D_, D_, 1, 0, 0);
  gemm_wmma_kernel<<<dim3(D_ / BN_, (B_ * S_) / BM_), gThr, 0, stream>>>(
      Zbf, Wv2T, bv2, nullptr, VT, B_ * S_, D_, D_, 2, 0, S_);

  // 6) cross flash attention (no mask)
  flash_attn_wmma_kernel<<<(B_ * T_ / 16) / 4, gThr, 0, stream>>>(
      Qb, Kbf, VT, attn, T_, S_, 0);

  // 7) LN2: y2 = LN(y1 + ca)
  ln_residual_kernel<<<M_, D_, 0, stream>>>(x1f, attn, g2, be2, x2f, x2bf);

  // 8) FFN
  gemm_wmma_kernel<<<dim3(F_ / BN_, M_ / BM_), gThr, 0, stream>>>(
      x2bf, W1T, b1, nullptr, hbuf, M_, F_, D_, 1, 1, 0);           // relu, bf16
  gemm_wmma_kernel<<<dim3(D_ / BN_, M_ / BM_), gThr, 0, stream>>>(
      hbuf, W2T, b2, attn, nullptr, M_, D_, F_, 0, 0, 0);           // f32

  // 9) LN3 -> d_out
  ln_residual_kernel<<<M_, D_, 0, stream>>>(x2f, attn, g3, be3, out, nullptr);
}
